// CapsNet_6236292513950
// MI455X (gfx1250) — compile-verified, matrix operand order checked by reading, not run on hardware
//
#include <hip/hip_runtime.h>
#include <hip/hip_bf16.h>
#include <math.h>

typedef __attribute__((ext_vector_type(16))) _Float16 v16h;
typedef __attribute__((ext_vector_type(8)))  _Float16 v8h;
typedef __attribute__((ext_vector_type(8)))  float    v8f;

#define NUM_ROUTES 1152
#define NUM_CAPS   10
#define CAP_DIM    16
#define PC_CAPS    8
#define BATCH      512

// ---------------------------------------------------------------------------
// f32 -> f16 conversion / zero / weight pad+convert to (N, Kp) row-major
// ---------------------------------------------------------------------------
__global__ void cvt_f32_to_f16(const float* __restrict__ in, _Float16* __restrict__ out, int n) {
    int i = blockIdx.x * blockDim.x + threadIdx.x;
    if (i < n) out[i] = (_Float16)in[i];
}

__global__ void zero_f32(float* __restrict__ p, int n) {
    int i = blockIdx.x * blockDim.x + threadIdx.x;
    if (i < n) p[i] = 0.0f;
}

__global__ void pad_weights_f16(const float* __restrict__ w, _Float16* __restrict__ out,
                                int N, int K, int Kp) {
    int i = blockIdx.x * blockDim.x + threadIdx.x;
    if (i >= N * Kp) return;
    int n = i / Kp, k = i - n * Kp;
    out[i] = (_Float16)((k < K) ? w[(size_t)n * K + k] : 0.0f);
}

// ---------------------------------------------------------------------------
// CDNA5 f16 WMMA fragment layout (V_WMMA_F32_16X16X32_F16):
//  A (16x32): lane holds row m = lane&15; halves = K[ 8*(lane>>4) .. +7 ]
//             and K[ 16 + 8*(lane>>4) .. +7 ]  -> two contiguous 8-half runs.
//  B (32x16): lane holds col n = lane&15; same K runs.
//  C/D (16x16 f32): element j: m = j + 8*(lane>>4), n = lane&15.
// ---------------------------------------------------------------------------
__device__ __forceinline__ v16h frag_concat(v8h lo, v8h hi) {
    return __builtin_shufflevector(lo, hi, 0, 1, 2, 3, 4, 5, 6, 7,
                                   8, 9, 10, 11, 12, 13, 14, 15);
}

// ---------------------------------------------------------------------------
// Implicit-GEMM 9x9 conv. M = B*Ho*Wo, N = Cout, K = Cin*81 (padded to Kp).
//  in : f16 (B,Cin,Hin,Win); wtp : f16 (Cout, Kp) row-major zero-padded.
// Block: 256 thr = 8 waves; block tile = 128(M) x 64(N):
//  each wave computes ONE M-tile x FOUR N-tiles (4 accumulators) so the A
//  fragment staged in LDS feeds 4 WMMAs (4x arithmetic intensity).
// Grid exact: M % 128 == 0, Cout % 64 == 0.
// A tile (128x32) staged in LDS (stager owns one K column -> 1 div set/step),
// consumed as 2x ds_load_b128; B read as 2x global_load_b128 per lane per tile.
// ---------------------------------------------------------------------------
__global__ void conv_wmma_kernel(const _Float16* __restrict__ in,
                                 const _Float16* __restrict__ wtp,
                                 const float* __restrict__ bias,
                                 float* __restrict__ outF, _Float16* __restrict__ outH,
                                 int Bn, int Cin, int Hin, int Win,
                                 int Cout, int Hout, int Wout, int stride,
                                 int relu, int Kp) {
    __shared__ __align__(16) _Float16 sA[128 * 32];

    const int HW = Hout * Wout;
    const int M  = Bn * HW;
    const int K  = Cin * 81;
    const int Mtiles = M >> 4;
    const int mblk = blockIdx.x % (Mtiles >> 3);
    const int nblk = blockIdx.x / (Mtiles >> 3);
    const int t    = threadIdx.x;
    const int wave = t >> 5;
    const int lane = t & 31;
    const int m0   = mblk * 128;
    const int n0   = nblk * 64;

    // ---- staging coordinates: this thread owns LDS column kk, rows mrow0+8i
    const int kk    = t & 31;
    const int mrow0 = t >> 5;
    int rowbase[16];
#pragma unroll
    for (int i = 0; i < 16; ++i) {
        int m  = m0 + mrow0 + i * 8;
        int bi = m / HW;
        int rm = m - bi * HW;
        int oy = rm / Wout;
        int ox = rm - oy * Wout;
        rowbase[i] = ((bi * Cin) * Hin + oy * stride) * Win + ox * stride;
    }

    // ---- consumer coordinates
    const int rowA  = wave * 16 + (lane & 15);       // local A row in LDS
    const int kchip = (lane >> 4) << 3;              // 0 or 8
    const _Float16* wrow0 = wtp + (size_t)(n0 + 0 * 16 + (lane & 15)) * Kp;
    const _Float16* wrow1 = wtp + (size_t)(n0 + 1 * 16 + (lane & 15)) * Kp;
    const _Float16* wrow2 = wtp + (size_t)(n0 + 2 * 16 + (lane & 15)) * Kp;
    const _Float16* wrow3 = wtp + (size_t)(n0 + 3 * 16 + (lane & 15)) * Kp;

    v8f acc0 = {0.f, 0.f, 0.f, 0.f, 0.f, 0.f, 0.f, 0.f};
    v8f acc1 = acc0, acc2 = acc0, acc3 = acc0;

    for (int k0 = 0; k0 < Kp; k0 += 32) {
        // decompose this thread's K column once per step
        int k = k0 + kk;
        int koffpart = 0;
        bool kin = (k < K);
        if (kin) {
            int ic = k / 81;
            int r9 = k - ic * 81;
            int ky = r9 / 9;
            int kx = r9 - ky * 9;
            koffpart = (ic * Hin + ky) * Win + kx;
        }
        __syncthreads();
#pragma unroll
        for (int i = 0; i < 16; ++i) {
            _Float16 val = (_Float16)0.f;
            if (kin) val = in[(size_t)(rowbase[i] + koffpart)];
            sA[(mrow0 + i * 8) * 32 + kk] = val;
        }
        __syncthreads();

        // prefetch next K chunk of the streamed weight rows (global_prefetch_b8)
        if (k0 + 32 < Kp) {
            __builtin_prefetch((const void*)(wrow0 + k0 + 32 + kchip), 0, 1);
            __builtin_prefetch((const void*)(wrow1 + k0 + 32 + kchip), 0, 1);
            __builtin_prefetch((const void*)(wrow2 + k0 + 32 + kchip), 0, 1);
            __builtin_prefetch((const void*)(wrow3 + k0 + 32 + kchip), 0, 1);
        }

        v8h a0 = *(const v8h*)&sA[rowA * 32 + kchip];
        v8h a1 = *(const v8h*)&sA[rowA * 32 + 16 + kchip];
        v16h af = frag_concat(a0, a1);

        v8h b0, b1;
        b0 = *(const v8h*)(wrow0 + k0 + kchip);
        b1 = *(const v8h*)(wrow0 + k0 + 16 + kchip);
        acc0 = __builtin_amdgcn_wmma_f32_16x16x32_f16(false, af, false, frag_concat(b0, b1),
                                                      (short)0, acc0, false, false);
        b0 = *(const v8h*)(wrow1 + k0 + kchip);
        b1 = *(const v8h*)(wrow1 + k0 + 16 + kchip);
        acc1 = __builtin_amdgcn_wmma_f32_16x16x32_f16(false, af, false, frag_concat(b0, b1),
                                                      (short)0, acc1, false, false);
        b0 = *(const v8h*)(wrow2 + k0 + kchip);
        b1 = *(const v8h*)(wrow2 + k0 + 16 + kchip);
        acc2 = __builtin_amdgcn_wmma_f32_16x16x32_f16(false, af, false, frag_concat(b0, b1),
                                                      (short)0, acc2, false, false);
        b0 = *(const v8h*)(wrow3 + k0 + kchip);
        b1 = *(const v8h*)(wrow3 + k0 + 16 + kchip);
        acc3 = __builtin_amdgcn_wmma_f32_16x16x32_f16(false, af, false, frag_concat(b0, b1),
                                                      (short)0, acc3, false, false);
    }

    const int tmg = mblk * 8 + wave;
#pragma unroll
    for (int jt = 0; jt < 4; ++jt) {
        v8f acc = (jt == 0) ? acc0 : (jt == 1) ? acc1 : (jt == 2) ? acc2 : acc3;
        int n = n0 + jt * 16 + (lane & 15);
        float bb = bias ? bias[n] : 0.0f;
#pragma unroll
        for (int j = 0; j < 8; ++j) {
            int mm = tmg * 16 + j + ((lane >> 4) << 3);
            float val = acc[j] + bb;
            if (relu) val = fmaxf(val, 0.0f);
            int bo = mm / HW, rr = mm - bo * HW, yy = rr / Wout, xx = rr - yy * Wout;
            size_t oi = (((size_t)bo * Cout + n) * Hout + yy) * Wout + xx;
            if (outH) outH[oi] = (_Float16)val;
            if (outF) outF[oi] = val;
        }
    }
}

// ---------------------------------------------------------------------------
// Generic WMMA GEMM: C = act(A @ B + bias). A f16 (trans opt), B f16 (K,N) ldb.
// Block: 8 waves = 8 M-tiles x 1 N-tile; requires M % 128 == 0, K % 32 == 0,
// lda % 8 == 0 when !transA. B staged transposed [n][k] in LDS; A direct
// global_load_b128 (!transA) or LDS-staged (transA). act: 0 none 1 relu 2 sigm.
// ---------------------------------------------------------------------------
__global__ void gemm_wmma_kernel(const _Float16* __restrict__ A,
                                 const _Float16* __restrict__ Bm,
                                 const float* __restrict__ bias,
                                 float* __restrict__ Cf, _Float16* __restrict__ Ch,
                                 int M, int N, int K, int lda, int ldb,
                                 int transA, int act) {
    __shared__ __align__(16) _Float16 sB[16 * 32];    // [n_local][k_local]
    __shared__ __align__(16) _Float16 sAt[128 * 32];  // used only when transA

    const int Mtiles = M >> 4;
    const int mblk = blockIdx.x % (Mtiles >> 3);
    const int tn   = blockIdx.x / (Mtiles >> 3);
    const int t    = threadIdx.x;
    const int wave = t >> 5;
    const int lane = t & 31;
    const int m0   = mblk * 128;
    const int n0   = tn * 16;

    // B staging: thread owns halves (2t, 2t+1): row r = t>>3, cols c, c+1
    const int rB = t >> 3;
    const int cB = (t & 7) * 2;
    // transA staging: thread owns K column kk, rows mrow0+8i
    const int kk    = t & 31;
    const int mrow0 = t >> 5;

    // consumer coords
    const int rowA  = wave * 16 + (lane & 15);
    const int mA    = m0 + rowA;
    const int nB    = n0 + (lane & 15);
    const int kchip = (lane >> 4) << 3;
    const _Float16* arow = A + (size_t)mA * lda;

    v8f acc = {0.f, 0.f, 0.f, 0.f, 0.f, 0.f, 0.f, 0.f};

    for (int k0 = 0; k0 < K; k0 += 32) {
        __syncthreads();
        // stage B tile (32 x 16) transposed
        {
            const _Float16* brow = Bm + (size_t)(k0 + rB) * ldb + n0;
            sB[(cB + 0) * 32 + rB] = brow[cB + 0];
            sB[(cB + 1) * 32 + rB] = brow[cB + 1];
        }
        if (transA) {
            int k = k0 + kk;
#pragma unroll
            for (int i = 0; i < 16; ++i) {
                int m = m0 + mrow0 + i * 8;
                sAt[(mrow0 + i * 8) * 32 + kk] = A[(size_t)k * lda + m];
            }
        }
        __syncthreads();

        v8h a0, a1;
        if (transA) {
            a0 = *(const v8h*)&sAt[rowA * 32 + kchip];
            a1 = *(const v8h*)&sAt[rowA * 32 + 16 + kchip];
        } else {
            if (k0 + 32 < K) __builtin_prefetch((const void*)(arow + k0 + 32 + kchip), 0, 1);
            a0 = *(const v8h*)(arow + k0 + kchip);
            a1 = *(const v8h*)(arow + k0 + 16 + kchip);
        }
        v8h b0 = *(const v8h*)&sB[(lane & 15) * 32 + kchip];
        v8h b1 = *(const v8h*)&sB[(lane & 15) * 32 + 16 + kchip];
        acc = __builtin_amdgcn_wmma_f32_16x16x32_f16(false, frag_concat(a0, a1),
                                                     false, frag_concat(b0, b1),
                                                     (short)0, acc, false, false);
    }

    const float bb = bias ? bias[nB] : 0.0f;
    const int tmg = mblk * 8 + wave;
#pragma unroll
    for (int j = 0; j < 8; ++j) {
        int mm = tmg * 16 + j + ((lane >> 4) << 3);
        float val = acc[j] + bb;
        if (act == 1) val = fmaxf(val, 0.0f);
        else if (act == 2) val = 1.0f / (1.0f + expf(-val));
        size_t oi = (size_t)mm * N + nB;
        if (Cf) Cf[oi] = val;
        if (Ch) Ch[oi] = (_Float16)val;
    }
}

// ---------------------------------------------------------------------------
// squash over primary capsules: u_pre (B,256,6,6) f32 == flat (B, 1152, 8)
// ---------------------------------------------------------------------------
__global__ void squash_u_kernel(const float* __restrict__ upre,
                                float* __restrict__ u, _Float16* __restrict__ uh) {
    int tid = blockIdx.x * blockDim.x + threadIdx.x;  // (b, r)
    if (tid >= BATCH * NUM_ROUTES) return;
    const float* p = upre + (size_t)tid * PC_CAPS;
    float sn = 1e-07f;
#pragma unroll
    for (int c = 0; c < PC_CAPS; ++c) sn += p[c] * p[c];
    float scale = sn / ((1.0f + sn) * sqrtf(sn));
#pragma unroll
    for (int c = 0; c < PC_CAPS; ++c) {
        float v = p[c] * scale;
        u[(size_t)tid * PC_CAPS + c]  = v;
        uh[(size_t)tid * PC_CAPS + c] = (_Float16)v;
    }
}

__global__ void squash_v_kernel(const float* __restrict__ s,
                                float* __restrict__ v, _Float16* __restrict__ vh) {
    int tid = blockIdx.x * blockDim.x + threadIdx.x;  // (b, k)
    if (tid >= BATCH * NUM_CAPS) return;
    const float* p = s + (size_t)tid * CAP_DIM;
    float sn = 0.0f;
#pragma unroll
    for (int o = 0; o < CAP_DIM; ++o) sn += p[o] * p[o];
    float scale = sn / ((1.0f + sn) * sqrtf(sn));
#pragma unroll
    for (int o = 0; o < CAP_DIM; ++o) {
        float x = p[o] * scale;
        v[(size_t)tid * CAP_DIM + o]  = x;
        vh[(size_t)tid * CAP_DIM + o] = (_Float16)x;
    }
}

// ---------------------------------------------------------------------------
// softmax over routes (axis 0); one block per object capsule k
// ---------------------------------------------------------------------------
__global__ void softmax_routes_kernel(const float* __restrict__ bij, float* __restrict__ cc) {
    __shared__ float red[256];
    const int k = blockIdx.x;
    const int t = threadIdx.x;
    float mx = -1e30f;
    for (int r = t; r < NUM_ROUTES; r += 256) mx = fmaxf(mx, bij[r * NUM_CAPS + k]);
    red[t] = mx; __syncthreads();
    for (int st = 128; st > 0; st >>= 1) { if (t < st) red[t] = fmaxf(red[t], red[t + st]); __syncthreads(); }
    float mxv = red[0]; __syncthreads();
    float sm = 0.0f;
    for (int r = t; r < NUM_ROUTES; r += 256) sm += expf(bij[r * NUM_CAPS + k] - mxv);
    red[t] = sm; __syncthreads();
    for (int st = 128; st > 0; st >>= 1) { if (t < st) red[t] += red[t + st]; __syncthreads(); }
    float inv = 1.0f / red[0];
    for (int r = t; r < NUM_ROUTES; r += 256)
        cc[r * NUM_CAPS + k] = expf(bij[r * NUM_CAPS + k] - mxv) * inv;
}

// Wct[(r*8+c), (k*16+o)] = f16( c[r,k] * W[r,k,o,c] )
__global__ void fold_w_kernel(const float* __restrict__ W, const float* __restrict__ cc,
                              _Float16* __restrict__ Wct) {
    int tid = blockIdx.x * blockDim.x + threadIdx.x;
    if (tid >= NUM_ROUTES * NUM_CAPS * CAP_DIM * PC_CAPS) return;
    int c = tid & 7;
    int o = (tid >> 3) & 15;
    int k = (tid / 128) % NUM_CAPS;
    int r = tid / 1280;
    Wct[(size_t)(r * PC_CAPS + c) * (NUM_CAPS * CAP_DIM) + k * CAP_DIM + o] =
        (_Float16)(cc[r * NUM_CAPS + k] * W[tid]);
}

// b_ij[r,k] += (1/B) * sum_{c,o} W[r,k,o,c] * T[(r*8+c), (k*16+o)]
__global__ void delta_kernel(const float* __restrict__ W, const float* __restrict__ T,
                             float* __restrict__ bij) {
    int tid = blockIdx.x * blockDim.x + threadIdx.x;  // (r, k)
    if (tid >= NUM_ROUTES * NUM_CAPS) return;
    int r = tid / NUM_CAPS, k = tid % NUM_CAPS;
    const float* Wp = W + (size_t)tid * CAP_DIM * PC_CAPS;
    float acc = 0.0f;
#pragma unroll
    for (int o = 0; o < CAP_DIM; ++o)
#pragma unroll
        for (int c = 0; c < PC_CAPS; ++c)
            acc += Wp[o * PC_CAPS + c] *
                   T[(size_t)(r * PC_CAPS + c) * (NUM_CAPS * CAP_DIM) + k * CAP_DIM + o];
    bij[tid] += acc * (1.0f / (float)BATCH);
}

// ---------------------------------------------------------------------------
// head: logits -> argmax -> mask / masked decoder input / obj_vectors output
// ---------------------------------------------------------------------------
__global__ void head_kernel(const float* __restrict__ v,
                            float* __restrict__ d_obj, float* __restrict__ d_mask,
                            _Float16* __restrict__ masked_h) {
    __shared__ float lg[NUM_CAPS];
    __shared__ int bestS;
    const int b = blockIdx.x;
    const int t = threadIdx.x;
    if (t < NUM_CAPS) {
        float s = 0.0f;
#pragma unroll
        for (int o = 0; o < CAP_DIM; ++o) {
            float x = v[((size_t)b * NUM_CAPS + t) * CAP_DIM + o];
            s += x * x;
        }
        lg[t] = sqrtf(s);
    }
    __syncthreads();
    if (t == 0) {
        int best = 0; float bv = lg[0];
        for (int k = 1; k < NUM_CAPS; ++k) if (lg[k] > bv) { bv = lg[k]; best = k; }
        bestS = best;
    }
    __syncthreads();
    if (t < NUM_CAPS) d_mask[(size_t)b * NUM_CAPS + t] = (t == bestS) ? 1.0f : 0.0f;
    for (int i = t; i < NUM_CAPS * CAP_DIM; i += 32) {
        float val = v[(size_t)b * NUM_CAPS * CAP_DIM + i];
        d_obj[(size_t)b * NUM_CAPS * CAP_DIM + i] = val;
        masked_h[(size_t)b * NUM_CAPS * CAP_DIM + i] =
            (_Float16)(((i >> 4) == bestS) ? val : 0.0f);
    }
}

// ---------------------------------------------------------------------------
extern "C" void kernel_launch(void* const* d_in, const int* in_sizes, int n_in,
                              void* d_out, int out_size, void* d_ws, size_t ws_size,
                              hipStream_t stream) {
    const float* image  = (const float*)d_in[0];
    const float* conv_w = (const float*)d_in[1];
    const float* conv_b = (const float*)d_in[2];
    const float* pc_w   = (const float*)d_in[3];
    const float* pc_b   = (const float*)d_in[4];
    const float* W_obj  = (const float*)d_in[5];
    const float* dec_w1 = (const float*)d_in[6];
    const float* dec_b1 = (const float*)d_in[7];
    const float* dec_w2 = (const float*)d_in[8];
    const float* dec_b2 = (const float*)d_in[9];
    const float* dec_w3 = (const float*)d_in[10];
    const float* dec_b3 = (const float*)d_in[11];

    char* ws = (char*)d_ws;
    _Float16* img_h   = (_Float16*)(ws + 0);            // 512*784
    _Float16* convw_p = (_Float16*)(ws + 802816);       // 256*96 (padded Kp=96)
    _Float16* pcw_h   = (_Float16*)(ws + 851968);       // 256*20736
    _Float16* x_h     = (_Float16*)(ws + 11468800);     // 512*256*400
    float*    upre    = (float*)   (ws + 116326400);    // 512*256*36
    float*    u_f     = (float*)   (ws + 135200768);    // 512*1152*8
    _Float16* u_h     = (_Float16*)(ws + 154075136);    // 512*9216
    float*    bij     = (float*)   (ws + 163512320);    // 1152*10
    float*    cc      = (float*)   (ws + 163558400);    // 1152*10
    _Float16* Wct     = (_Float16*)(ws + 163604480);    // 9216*160
    float*    s_f     = (float*)   (ws + 166553600);    // 512*160
    float*    v_f     = (float*)   (ws + 166881280);    // 512*160
    _Float16* v_h     = (_Float16*)(ws + 167208960);    // 512*160
    float*    T_f     = (float*)   (ws + 167372800);    // 9216*160
    _Float16* masked  = (_Float16*)(ws + 173271040);    // 512*160
    _Float16* h1_h    = (_Float16*)(ws + 173434880);    // 512*512
    _Float16* h2_h    = (_Float16*)(ws + 173959168);    // 512*1024
    _Float16* w1_h    = (_Float16*)(ws + 175007744);    // 160*512
    _Float16* w2_h    = (_Float16*)(ws + 175171584);    // 512*1024
    _Float16* w3_h    = (_Float16*)(ws + 176220160);    // 1024*784

    float* out_obj  = (float*)d_out;                           // (512,10,16,1)
    float* out_rec  = (float*)d_out + 512 * 160;               // (512,1,28,28)
    float* out_mask = (float*)d_out + 512 * 160 + 512 * 784;   // (512,10)

    auto cdiv = [](int a, int b) { return (a + b - 1) / b; };

    // --- precision conversions / weight reformat ---
    cvt_f32_to_f16<<<cdiv(512 * 784, 256), 256, 0, stream>>>(image, img_h, 512 * 784);
    pad_weights_f16<<<cdiv(256 * 96, 256), 256, 0, stream>>>(conv_w, convw_p, 256, 81, 96);
    pad_weights_f16<<<cdiv(256 * 20736, 256), 256, 0, stream>>>(pc_w, pcw_h, 256, 20736, 20736);
    cvt_f32_to_f16<<<cdiv(160 * 512, 256), 256, 0, stream>>>(dec_w1, w1_h, 160 * 512);
    cvt_f32_to_f16<<<cdiv(512 * 1024, 256), 256, 0, stream>>>(dec_w2, w2_h, 512 * 1024);
    cvt_f32_to_f16<<<cdiv(1024 * 784, 256), 256, 0, stream>>>(dec_w3, w3_h, 1024 * 784);

    // --- conv1: M=204800, N=256, K=81 (Kp=96), relu -> f16
    {
        int Mtiles = (512 * 400) / 16, Ntiles = 256 / 16;
        conv_wmma_kernel<<<(Mtiles / 8) * (Ntiles / 4), 256, 0, stream>>>(
            img_h, convw_p, conv_b, nullptr, x_h,
            512, 1, 28, 28, 256, 20, 20, 1, 1, 96);
    }
    // --- PC conv: M=18432, N=256, K=20736 -> f32
    {
        int Mtiles = (512 * 36) / 16, Ntiles = 256 / 16;
        conv_wmma_kernel<<<(Mtiles / 8) * (Ntiles / 4), 256, 0, stream>>>(
            x_h, pcw_h, pc_b, upre, nullptr,
            512, 256, 20, 20, 256, 6, 6, 2, 0, 20736);
    }
    squash_u_kernel<<<cdiv(512 * NUM_ROUTES, 256), 256, 0, stream>>>(upre, u_f, u_h);

    // --- dynamic routing (3 iters); u_hat never materialized ---
    zero_f32<<<cdiv(NUM_ROUTES * NUM_CAPS, 256), 256, 0, stream>>>(bij, NUM_ROUTES * NUM_CAPS);
    for (int it = 0; it < 3; ++it) {
        softmax_routes_kernel<<<NUM_CAPS, 256, 0, stream>>>(bij, cc);
        fold_w_kernel<<<cdiv(NUM_ROUTES * NUM_CAPS * CAP_DIM * PC_CAPS, 256), 256, 0, stream>>>(
            W_obj, cc, Wct);
        // s[b,ko] = u[b,rc] @ Wct[rc,ko] : M=512, N=160, K=9216
        gemm_wmma_kernel<<<(512 / 128) * (160 / 16), 256, 0, stream>>>(
            u_h, Wct, nullptr, s_f, nullptr, 512, 160, 9216, 9216, 160, 0, 0);
        squash_v_kernel<<<cdiv(512 * NUM_CAPS, 256), 256, 0, stream>>>(s_f, v_f, v_h);
        if (it < 2) {
            // T[rc,ko] = sum_b u[b,rc]*v[b,ko] : M=9216, N=160, K=512 (A^T)
            gemm_wmma_kernel<<<(9216 / 128) * (160 / 16), 256, 0, stream>>>(
                u_h, v_h, nullptr, T_f, nullptr, 9216, 160, 512, 9216, 160, 1, 0);
            delta_kernel<<<cdiv(NUM_ROUTES * NUM_CAPS, 256), 256, 0, stream>>>(W_obj, T_f, bij);
        }
    }

    head_kernel<<<512, 32, 0, stream>>>(v_f, out_obj, out_mask, masked);

    // --- decoder MLP ---
    gemm_wmma_kernel<<<(512 / 128) * (512 / 16), 256, 0, stream>>>(
        masked, w1_h, dec_b1, nullptr, h1_h, 512, 512, 160, 160, 512, 0, 1);
    gemm_wmma_kernel<<<(512 / 128) * (1024 / 16), 256, 0, stream>>>(
        h1_h, w2_h, dec_b2, nullptr, h2_h, 512, 1024, 512, 512, 1024, 0, 1);
    gemm_wmma_kernel<<<(512 / 128) * (784 / 16), 256, 0, stream>>>(
        h2_h, w3_h, dec_b3, out_rec, nullptr, 512, 784, 1024, 1024, 784, 0, 2);

    (void)in_sizes; (void)n_in; (void)out_size; (void)ws_size;
}
